// CourseRecommendationGNN_75273596830171
// MI455X (gfx1250) — matrix operand order repaired and checked by reading, not compiled, not run on hardware
//
#include <hip/hip_runtime.h>
#include <hip/hip_bf16.h>

typedef __attribute__((ext_vector_type(16))) __bf16 v16bf;
typedef __attribute__((ext_vector_type(8)))  float  v8f;

// ---------------------------------------------------------------------------
// Zero an int buffer (grid-stride). 0x00000000 == 0.0f, so usable for floats.
// ---------------------------------------------------------------------------
__global__ void zero_i32(int* __restrict__ p, size_t n) {
  size_t i = (size_t)blockIdx.x * blockDim.x + threadIdx.x;
  const size_t stride = (size_t)gridDim.x * blockDim.x;
  for (; i < n; i += stride) p[i] = 0;
}

// ---------------------------------------------------------------------------
// CSR build step 1: per-destination edge counts (int atomics, ~E total ops).
// ---------------------------------------------------------------------------
__global__ void count_kernel(const long long* __restrict__ dst,
                             int* __restrict__ cnt, int E) {
  const int e = blockIdx.x * blockDim.x + threadIdx.x;
  if (e < E) atomicAdd(&cnt[dst[e]], 1);
}

// ---------------------------------------------------------------------------
// CSR build step 2: single-block exclusive scan over N counts (N=50k ->
// 49 chunks of 1024, LDS ladder scan). Writes rowptr[0..N] and cursor[i]
// (= exclusive prefix, the fill cursor).
// ---------------------------------------------------------------------------
__global__ void scan_kernel(const int* __restrict__ cnt,
                            int* __restrict__ rowptr,
                            int* __restrict__ cursor, int n) {
  __shared__ int buf[1024];
  __shared__ int carry;
  if (threadIdx.x == 0) { carry = 0; rowptr[0] = 0; }
  __syncthreads();
  for (int base = 0; base < n; base += 1024) {
    const int i = base + (int)threadIdx.x;
    const int v = (i < n) ? cnt[i] : 0;
    buf[threadIdx.x] = v;
    __syncthreads();
#pragma unroll
    for (int off = 1; off < 1024; off <<= 1) {
      const int t = (threadIdx.x >= (unsigned)off) ? buf[threadIdx.x - off] : 0;
      __syncthreads();
      buf[threadIdx.x] += t;
      __syncthreads();
    }
    const int c = carry;
    if (i < n) {
      const int inc = buf[threadIdx.x];
      rowptr[i + 1] = c + inc;
      cursor[i]     = c + inc - v;
    }
    __syncthreads();
    if (threadIdx.x == 0) carry = c + buf[1023];
    __syncthreads();
  }
}

// ---------------------------------------------------------------------------
// CSR build step 3: bucket source indices by destination.
// ---------------------------------------------------------------------------
__global__ void fill_kernel(const long long* __restrict__ src,
                            const long long* __restrict__ dst,
                            int* __restrict__ cursor,
                            int* __restrict__ eidx, int E) {
  const int e = blockIdx.x * blockDim.x + threadIdx.x;
  if (e >= E) return;
  const int d = (int)dst[e];
  const int pos = atomicAdd(&cursor[d], 1);
  eidx[pos] = (int)src[e];
}

// ---------------------------------------------------------------------------
// Mean aggregation as gather-reduce: one wave per destination node, 32 lanes
// x float4 = 128-float accumulator in registers. Each edge costs one fully
// coalesced 512B row read (the unavoidable 410 MB of gather traffic at the
// 23.3 TB/s roofline); one normalized 512B store per node. No float atomics.
// ---------------------------------------------------------------------------
__global__ void gather_mean_128(const float* __restrict__ feat,
                                const int* __restrict__ rowptr,
                                const int* __restrict__ eidx,
                                float* __restrict__ agg, int nnodes) {
  const int lane = threadIdx.x & 31;
  const int v = blockIdx.x * (blockDim.x >> 5) + (threadIdx.x >> 5);
  if (v >= nnodes) return;
  const int beg = rowptr[v];
  const int end = rowptr[v + 1];
  float4 acc = {0.0f, 0.0f, 0.0f, 0.0f};
  for (int j = beg; j < end; ++j) {
    const int s = eidx[j];  // wave-uniform
    const float4 t = *(const float4*)(feat + (size_t)s * 128 + lane * 4);
    acc.x += t.x; acc.y += t.y; acc.z += t.z; acc.w += t.w;
  }
  const float inv = 1.0f / fmaxf((float)(end - beg), 1.0f);
  acc.x *= inv; acc.y *= inv; acc.z *= inv; acc.w *= inv;
  *(float4*)(agg + (size_t)v * 128 + lane * 4) = acc;
}

// ---------------------------------------------------------------------------
// Fused SAGE layer GEMM:  out = act( A0 @ Wl + A1 @ Wr + bias )
// A0 = pre-normalized mean aggregation, A1 = self features. K = 128.
// One wave computes a 16x16 output tile with v_wmma_f32_16x16x32_bf16
// (bf16 inputs, f32 accumulate). blockDim.x = (DN/16)*32.
//
// Fragment layouts as loaded here (A and B use the same contiguous k order):
//  A (16x32, 16-bit): lane l in [0,16): row M=l; elements 0..7 -> K = hi*16+0..7,
//                     elements 8..15 -> K = hi*16+8..15  (hi = lane>>4)
//  B (32x16, 16-bit): lane = column; lanes 0-15 hold K=0..15 (elem i -> K=i),
//                     lanes 16-31 hold K=16..31
//  C/D (16x16 f32):   lane = column; vgpr r -> row M = hi*8 + r
// ---------------------------------------------------------------------------
template <int DN, bool RELU>
__global__ void sage_gemm(const float* __restrict__ A0,    // [N x 128]
                          const float* __restrict__ A1,    // [N x 128]
                          const float* __restrict__ Wl,    // [128 x DN]
                          const float* __restrict__ Wr,    // [128 x DN]
                          const float* __restrict__ bias,  // [DN]
                          float* __restrict__ out,         // [N x DN]
                          int nrows) {
  constexpr int K = 128;
  const int lane = threadIdx.x & 31;
  const int wave = threadIdx.x >> 5;   // 0 .. DN/16-1
  const int l    = lane & 15;
  const int hi   = lane >> 4;          // 0 or 1
  const int m0   = blockIdx.x * 16;
  const int n0   = wave * 16;

  // Clamp row so EXEC stays all-1s (WMMA requirement); stores are guarded.
  int arow = m0 + l;
  if (arow > nrows - 1) arow = nrows - 1;
  const int ncol = n0 + l;

  v8f c = {};
#pragma unroll
  for (int pass = 0; pass < 2; ++pass) {
    const float* __restrict__ A = pass ? A1 : A0;
    const float* __restrict__ W = pass ? Wr : Wl;
#pragma unroll
    for (int k0 = 0; k0 < K; k0 += 32) {
      v16bf a, b;
      const float* ap = A + (size_t)arow * K + k0 + hi * 16;
#pragma unroll
      for (int i = 0; i < 8; ++i) {
        a[i]     = (__bf16)ap[i];
        a[8 + i] = (__bf16)ap[8 + i];
      }
      const float* wp = W + (size_t)(k0 + hi * 16) * DN + ncol;
#pragma unroll
      for (int i = 0; i < 16; ++i) {
        b[i] = (__bf16)wp[(size_t)i * DN];
      }
      c = __builtin_amdgcn_wmma_f32_16x16x32_bf16(
          /*neg_a=*/false, a, /*neg_b=*/false, b,
          /*c_mod=*/(short)0, c, /*reuse_a=*/false, /*reuse_b=*/false);
    }
  }

  const float bn = bias[ncol];
#pragma unroll
  for (int r = 0; r < 8; ++r) {
    const int row = m0 + hi * 8 + r;
    float v = c[r] + bn;
    if (RELU) v = fmaxf(v, 0.0f);
    if (row < nrows) out[(size_t)row * DN + ncol] = v;
  }
}

// ---------------------------------------------------------------------------
// Launcher.  Inputs (setup_inputs order):
//   0: x [N x 128] f32      1: edge_index [2 x E] i64
//   2: W_l1 [128x128]  3: W_r1 [128x128]  4: b1 [128]
//   5: W_l2 [128x64]   6: W_r2 [128x64]   7: b2 [64]
// Output: [N x 64] f32
// Workspace: cnt[N] | rowptr[N+1] | cursor[N] | eidx[E] | agg[N*128] | h[N*128]
// ---------------------------------------------------------------------------
extern "C" void kernel_launch(void* const* d_in, const int* in_sizes, int n_in,
                              void* d_out, int out_size, void* d_ws, size_t ws_size,
                              hipStream_t stream) {
  const float*     x    = (const float*)d_in[0];
  const long long* ei   = (const long long*)d_in[1];
  const float*     W_l1 = (const float*)d_in[2];
  const float*     W_r1 = (const float*)d_in[3];
  const float*     b1   = (const float*)d_in[4];
  const float*     W_l2 = (const float*)d_in[5];
  const float*     W_r2 = (const float*)d_in[6];
  const float*     b2   = (const float*)d_in[7];
  float*           out  = (float*)d_out;

  const int N = in_sizes[0] / 128;
  const int E = in_sizes[1] / 2;
  const long long* src = ei;
  const long long* dst = ei + E;

  int*   cnt    = (int*)d_ws;
  int*   rowptr = cnt + N;
  int*   cursor = rowptr + (N + 1);
  int*   eidx   = cursor + N;
  float* agg    = (float*)(eidx + ((E + 3) & ~3));  // 16B-align
  float* h      = agg + (size_t)N * 128;

  const int mblocks = (N + 15) / 16;  // N = 50000 -> 3125, exact 16-row tiles

  // ---- CSR build (once, reused by both layers) ----
  zero_i32<<<256, 256, 0, stream>>>(cnt, (size_t)N);
  count_kernel<<<(E + 255) / 256, 256, 0, stream>>>(dst, cnt, E);
  scan_kernel<<<1, 1024, 0, stream>>>(cnt, rowptr, cursor, N);
  fill_kernel<<<(E + 255) / 256, 256, 0, stream>>>(src, dst, cursor, eidx, E);

  // ---- layer 1 ----
  gather_mean_128<<<(N + 7) / 8, 256, 0, stream>>>(x, rowptr, eidx, agg, N);
  sage_gemm<128, true><<<mblocks, 256, 0, stream>>>(agg, x, W_l1, W_r1, b1, h, N);

  // ---- layer 2 (reuse agg buffer) ----
  gather_mean_128<<<(N + 7) / 8, 256, 0, stream>>>(h, rowptr, eidx, agg, N);
  sage_gemm<64, false><<<mblocks, 128, 0, stream>>>(agg, h, W_l2, W_r2, b2, out, N);
}
